// TitanMini_14757507629545
// MI455X (gfx1250) — matrix operand order repaired and checked by reading, not compiled, not run on hardware
//
#include <hip/hip_runtime.h>
#include <stdint.h>

// ---------------- problem constants ----------------
#define LAYERS 13
#define DMODEL 512
#define NHEAD  8
#define DKHEAD 64
#define DFFN   1920
#define SEQT   64
#define BATCH  512
#define MTOT   (BATCH * SEQT)   // 32768 rows
#define NREL   15               // 2*MAX_REL+1

typedef __attribute__((ext_vector_type(16))) __bf16 v16bf;
typedef __attribute__((ext_vector_type(8)))  __bf16 v8bf;
typedef __attribute__((ext_vector_type(8)))  float  v8f;

// float -> bf16 round-to-nearest-even
static __device__ inline __bf16 f2bf(float f) {
    union { float f; uint32_t u; } x; x.f = f;
    uint32_t r = x.u + 0x7FFFu + ((x.u >> 16) & 1u);
    uint16_t h = (uint16_t)(r >> 16);
    __bf16 o;
    __builtin_memcpy(&o, &h, sizeof(o));
    return o;
}

static __device__ inline v8f wmma_bf16(v16bf a, v16bf b, v8f c) {
    return __builtin_amdgcn_wmma_f32_16x16x32_bf16(false, a, false, b, (short)0, c,
                                                   false, false);
}

// A-fragment, 16x32 bf16 (row-major LDS, stride mult of 8).
// Lanes 0-15: row=lane, K={0..7,16..23}; lanes 16-31: row=lane-16, K={8..15,24..31}.
static __device__ inline v16bf load_frag_a(const __bf16* p, int stride, int lane) {
    int row = lane & 15;
    int k0  = (lane < 16) ? 0 : 8;
    v8bf lo = *reinterpret_cast<const v8bf*>(p + row * stride + k0);
    v8bf hi = *reinterpret_cast<const v8bf*>(p + row * stride + k0 + 16);
    return __builtin_shufflevector(lo, hi, 0,1,2,3,4,5,6,7,8,9,10,11,12,13,14,15);
}

// B-fragment, 32x16 bf16, from B^T (N rows x K cols) row-major LDS tile.
static __device__ inline v16bf load_frag_b(const __bf16* p, int stride, int lane) {
    int n  = lane & 15;
    int k0 = (lane < 16) ? 0 : 8;
    v8bf lo = *reinterpret_cast<const v8bf*>(p + n * stride + k0);
    v8bf hi = *reinterpret_cast<const v8bf*>(p + n * stride + k0 + 16);
    return __builtin_shufflevector(lo, hi, 0,1,2,3,4,5,6,7,8,9,10,11,12,13,14,15);
}

// ---- CDNA5 async global->LDS copy (16 bytes per lane), tracked by ASYNCcnt ----
static __device__ inline void async_ld16(const __bf16* gptr, __bf16* lptr) {
    uint32_t l = (uint32_t)(uintptr_t)lptr;   // flat addr low 32 = LDS offset
    uint64_t g = (uint64_t)(uintptr_t)gptr;
    asm volatile("global_load_async_to_lds_b128 %0, %1, off"
                 :: "v"(l), "v"(g) : "memory");
}
static __device__ inline void wait_async0() {
    asm volatile("s_wait_asynccnt 0" ::: "memory");
}

// ---------------- utility kernels ----------------
__global__ __launch_bounds__(256) void copy_f32_kernel(const float* __restrict__ src,
                                                       float* __restrict__ dst, size_t n) {
    size_t i = (size_t)blockIdx.x * 256 + threadIdx.x;
    if (i < n) dst[i] = src[i];
}

// W (K x N, f32 row-major) -> WT (N x K, bf16)
__global__ __launch_bounds__(256) void transpose_bf16_kernel(const float* __restrict__ src,
                                                             __bf16* __restrict__ dst,
                                                             int K, int N) {
    size_t idx = (size_t)blockIdx.x * 256 + threadIdx.x;
    if (idx < (size_t)K * N) {
        int k = (int)(idx / N);
        int n = (int)(idx % N);
        dst[(size_t)n * K + k] = f2bf(src[idx]);
    }
}

// LayerNorm over D=512, one row per 128-thread block, f32 in -> bf16 out
__global__ __launch_bounds__(128) void layernorm_kernel(const float* __restrict__ x,
                                                        const float* __restrict__ w,
                                                        const float* __restrict__ b,
                                                        __bf16* __restrict__ out) {
    int row = blockIdx.x;
    int tid = threadIdx.x;
    __shared__ float r1[128];
    __shared__ float r2[128];
    float vals[4];
    float s1 = 0.f, s2 = 0.f;
#pragma unroll
    for (int i = 0; i < 4; ++i) {
        float v = x[(size_t)row * DMODEL + tid + i * 128];
        vals[i] = v; s1 += v; s2 += v * v;
    }
    r1[tid] = s1; r2[tid] = s2;
    __syncthreads();
    for (int off = 64; off > 0; off >>= 1) {
        if (tid < off) { r1[tid] += r1[tid + off]; r2[tid] += r2[tid + off]; }
        __syncthreads();
    }
    float mean = r1[0] * (1.f / DMODEL);
    float var  = r2[0] * (1.f / DMODEL) - mean * mean;
    float rs   = rsqrtf(var + 1e-5f);
#pragma unroll
    for (int i = 0; i < 4; ++i) {
        int c = tid + i * 128;
        out[(size_t)row * DMODEL + c] = f2bf((vals[i] - mean) * rs * w[c] + b[c]);
    }
}

// ---------------- WMMA GEMM, 256 threads, 128x64 tile, K-step 64, async double-buffer ----
// C(MxN) = A(MxK) @ BT(NxK)^T + bias.  MODE 0: bf16 store. MODE 1: f32 residual +=.
template <int MODE>
__global__ __launch_bounds__(256) void gemm_wmma_kernel(const __bf16* __restrict__ A, int lda,
                                                        const __bf16* __restrict__ BT, int ldb,
                                                        const float* __restrict__ bias,
                                                        __bf16* __restrict__ Cbf,
                                                        float* __restrict__ Xres, int ldc,
                                                        int K) {
    __shared__ __align__(16) __bf16 As[2][128 * 64];
    __shared__ __align__(16) __bf16 Bs[2][64 * 64];
    const int tid  = threadIdx.x;
    const int lane = tid & 31;
    const int wave = tid >> 5;              // 0..7 -> 16-row strip of 128
    const int mBase = blockIdx.y * 128;
    const int nBase = blockIdx.x * 64;

    v8f acc[4] = {};

    auto stage = [&](int buf, int kk) {
        // A: 128x64 bf16 = 1024 16B chunks; 4 per thread
#pragma unroll
        for (int t = 0; t < 4; ++t) {
            int cidx = tid + t * 256;
            int row  = cidx >> 3;
            int c8   = (cidx & 7) * 8;
            async_ld16(A + (size_t)(mBase + row) * lda + kk + c8,
                       &As[buf][row * 64 + c8]);
        }
        // B: 64x64 bf16 = 512 chunks; 2 per thread
#pragma unroll
        for (int t = 0; t < 2; ++t) {
            int cidx = tid + t * 256;
            int row  = cidx >> 3;
            int c8   = (cidx & 7) * 8;
            async_ld16(BT + (size_t)(nBase + row) * ldb + kk + c8,
                       &Bs[buf][row * 64 + c8]);
        }
    };

    const int nsteps = K >> 6;   // K-step 64
    int buf = 0;
    stage(0, 0);
    wait_async0();
    __syncthreads();

    for (int s = 0; s < nsteps; ++s) {
        if (s + 1 < nsteps) stage(buf ^ 1, (s + 1) << 6);
#pragma unroll
        for (int kc = 0; kc < 64; kc += 32) {
            v16bf af = load_frag_a(&As[buf][wave * 16 * 64 + kc], 64, lane);
#pragma unroll
            for (int nt = 0; nt < 4; ++nt) {
                v16bf bfr = load_frag_b(&Bs[buf][nt * 16 * 64 + kc], 64, lane);
                acc[nt] = wmma_bf16(af, bfr, acc[nt]);
            }
        }
        wait_async0();
        __syncthreads();
        buf ^= 1;
    }

    const int halfsel = lane >> 4;
    const int nIn     = lane & 15;
#pragma unroll
    for (int nt = 0; nt < 4; ++nt) {
        int col = nBase + nt * 16 + nIn;
        float bv = bias[col];
#pragma unroll
        for (int v = 0; v < 8; ++v) {
            int row = mBase + wave * 16 + v + 8 * halfsel;
            float val = acc[nt][v] + bv;
            if (MODE == 0) Cbf[(size_t)row * ldc + col] = f2bf(val);
            else           Xres[(size_t)row * ldc + col] += val;
        }
    }
}

// ---------------- fused attention per (b, head) ----------------
__global__ __launch_bounds__(128) void attention_kernel(const __bf16* __restrict__ q,
                                                        const __bf16* __restrict__ k,
                                                        const __bf16* __restrict__ v,
                                                        const float* __restrict__ tab, // 15*15*H
                                                        __bf16* __restrict__ ctx) {
    __shared__ __align__(16) __bf16 Qs[64 * 64];
    __shared__ __align__(16) __bf16 Ks[64 * 64];
    __shared__ __align__(16) __bf16 VTs[64 * 64];  // [d][s]
    __shared__ __align__(16) __bf16 Ps[64 * 64];
    __shared__ float Ss[64 * 65];

    int bh = blockIdx.x;
    int b  = bh >> 3;
    int h  = bh & 7;
    int tid  = threadIdx.x;
    int lane = tid & 31;
    int wave = tid >> 5;

    size_t rowBase = (size_t)b * SEQT;
    int ch = h * DKHEAD;

    int r   = tid >> 1;
    int seg = (tid & 1) * 32;
#pragma unroll
    for (int i = 0; i < 4; ++i) {
        int c = seg + i * 8;
        v8bf qv = *reinterpret_cast<const v8bf*>(q + (rowBase + r) * DMODEL + ch + c);
        *reinterpret_cast<v8bf*>(Qs + r * 64 + c) = qv;
        v8bf kv = *reinterpret_cast<const v8bf*>(k + (rowBase + r) * DMODEL + ch + c);
        *reinterpret_cast<v8bf*>(Ks + r * 64 + c) = kv;
        v8bf vv = *reinterpret_cast<const v8bf*>(v + (rowBase + r) * DMODEL + ch + c);
#pragma unroll
        for (int e = 0; e < 8; ++e) VTs[(c + e) * 64 + r] = vv[e];
    }
    __syncthreads();

    // scores = q @ k^T
    v8f sc[4] = {};
#pragma unroll
    for (int kc = 0; kc < DKHEAD; kc += 32) {
        v16bf af = load_frag_a(Qs + wave * 16 * 64 + kc, 64, lane);
#pragma unroll
        for (int nt = 0; nt < 4; ++nt) {
            v16bf bfr = load_frag_b(Ks + nt * 16 * 64 + kc, 64, lane);
            sc[nt] = wmma_bf16(af, bfr, sc[nt]);
        }
    }
    int halfsel = lane >> 4;
    int nIn     = lane & 15;
#pragma unroll
    for (int nt = 0; nt < 4; ++nt) {
        int s = nt * 16 + nIn;
#pragma unroll
        for (int vv = 0; vv < 8; ++vv) {
            int t = wave * 16 + vv + 8 * halfsel;
            int dr = (t >> 3) - (s >> 3); dr = min(7, max(-7, dr)) + 7;
            int df = (t & 7) - (s & 7);   df = min(7, max(-7, df)) + 7;
            float bias = 2.f * tanhf(tab[(dr * NREL + df) * NHEAD + h]);
            float val = sc[nt][vv] * 0.125f + bias;      // 1/sqrt(64)
            val = fminf(50.f, fmaxf(-50.f, val));
            Ss[t * 65 + s] = val;
        }
    }
    __syncthreads();

    if (tid < 64) {
        float mx = -1e30f;
        for (int s = 0; s < 64; ++s) mx = fmaxf(mx, Ss[tid * 65 + s]);
        float sum = 0.f;
        for (int s = 0; s < 64; ++s) {
            float e = __expf(Ss[tid * 65 + s] - mx);
            Ss[tid * 65 + s] = e;
            sum += e;
        }
        float inv = 1.f / sum;
        for (int s = 0; s < 64; ++s) Ps[tid * 64 + s] = f2bf(Ss[tid * 65 + s] * inv);
    }
    __syncthreads();

    // ctx = P @ V
    v8f cc[4] = {};
#pragma unroll
    for (int kc = 0; kc < SEQT; kc += 32) {
        v16bf af = load_frag_a(Ps + wave * 16 * 64 + kc, 64, lane);
#pragma unroll
        for (int nt = 0; nt < 4; ++nt) {
            v16bf bfr = load_frag_b(VTs + nt * 16 * 64 + kc, 64, lane);
            cc[nt] = wmma_bf16(af, bfr, cc[nt]);
        }
    }
#pragma unroll
    for (int nt = 0; nt < 4; ++nt) {
        int d = nt * 16 + nIn;
#pragma unroll
        for (int vv = 0; vv < 8; ++vv) {
            int t = wave * 16 + vv + 8 * halfsel;
            ctx[(rowBase + t) * DMODEL + ch + d] = f2bf(cc[nt][vv]);
        }
    }
}

// ---- fused gated FFN: G = gelu(h@Wp1+b1)*(h@Wp2+b2), 256 thr, 128x64 tile, K-step 32 ----
__global__ __launch_bounds__(256) void ffn_gate_kernel(const __bf16* __restrict__ A,   // M x D
                                                       const __bf16* __restrict__ WpT, // 2DFF x D
                                                       const float* __restrict__ bp,   // 2DFF
                                                       __bf16* __restrict__ G) {       // M x DFF
    __shared__ __align__(16) __bf16 As[2][128 * 32];
    __shared__ __align__(16) __bf16 B1s[2][64 * 32];
    __shared__ __align__(16) __bf16 B2s[2][64 * 32];
    const int tid  = threadIdx.x;
    const int lane = tid & 31;
    const int wave = tid >> 5;
    const int mBase = blockIdx.y * 128;
    const int nBase = blockIdx.x * 64;

    v8f a1[4] = {};
    v8f a2[4] = {};

    auto stage = [&](int buf, int kk) {
        // A: 128x32 = 512 chunks; 2 per thread
#pragma unroll
        for (int t = 0; t < 2; ++t) {
            int cidx = tid + t * 256;
            int row  = cidx >> 2;
            int c8   = (cidx & 3) * 8;
            async_ld16(A + (size_t)(mBase + row) * DMODEL + kk + c8,
                       &As[buf][row * 32 + c8]);
        }
        // B1/B2: 64x32 = 256 chunks each; 1 per thread
        {
            int row = tid >> 2;
            int c8  = (tid & 3) * 8;
            async_ld16(WpT + (size_t)(nBase + row) * DMODEL + kk + c8,
                       &B1s[buf][row * 32 + c8]);
            async_ld16(WpT + (size_t)(DFFN + nBase + row) * DMODEL + kk + c8,
                       &B2s[buf][row * 32 + c8]);
        }
    };

    const int nsteps = DMODEL / 32;
    int buf = 0;
    stage(0, 0);
    wait_async0();
    __syncthreads();

    for (int s = 0; s < nsteps; ++s) {
        if (s + 1 < nsteps) stage(buf ^ 1, (s + 1) * 32);
        v16bf af = load_frag_a(&As[buf][wave * 16 * 32], 32, lane);
#pragma unroll
        for (int nt = 0; nt < 4; ++nt) {
            v16bf bf1 = load_frag_b(&B1s[buf][nt * 16 * 32], 32, lane);
            a1[nt] = wmma_bf16(af, bf1, a1[nt]);
            v16bf bf2 = load_frag_b(&B2s[buf][nt * 16 * 32], 32, lane);
            a2[nt] = wmma_bf16(af, bf2, a2[nt]);
        }
        wait_async0();
        __syncthreads();
        buf ^= 1;
    }

    const int halfsel = lane >> 4;
    const int nIn     = lane & 15;
#pragma unroll
    for (int nt = 0; nt < 4; ++nt) {
        int col = nBase + nt * 16 + nIn;
        float b1 = bp[col];
        float b2 = bp[DFFN + col];
#pragma unroll
        for (int v = 0; v < 8; ++v) {
            int row = mBase + wave * 16 + v + 8 * halfsel;
            float x1 = a1[nt][v] + b1;
            float x2 = a2[nt][v] + b2;
            float ge = 0.5f * x1 * (1.f + erff(x1 * 0.70710678118654752f));
            G[(size_t)row * DFFN + col] = f2bf(ge * x2);
        }
    }
}

// ---------------- host orchestration ----------------
extern "C" void kernel_launch(void* const* d_in, const int* in_sizes, int n_in,
                              void* d_out, int out_size, void* d_ws, size_t ws_size,
                              hipStream_t stream) {
    (void)in_sizes; (void)n_in; (void)out_size; (void)ws_size;

    const float* x_in  = (const float*)d_in[0];
    const float* ln1w  = (const float*)d_in[1];
    const float* ln1b  = (const float*)d_in[2];
    const float* Wq    = (const float*)d_in[3];
    const float* bq    = (const float*)d_in[4];
    const float* Wk    = (const float*)d_in[5];
    const float* bk    = (const float*)d_in[6];
    const float* Wv    = (const float*)d_in[7];
    const float* bv    = (const float*)d_in[8];
    const float* Wo    = (const float*)d_in[9];
    const float* bo    = (const float*)d_in[10];
    const float* tab   = (const float*)d_in[11];
    const float* ln2w  = (const float*)d_in[12];
    const float* ln2b  = (const float*)d_in[13];
    const float* Wp    = (const float*)d_in[14];
    const float* bp    = (const float*)d_in[15];
    const float* Wf    = (const float*)d_in[16];
    const float* bfb   = (const float*)d_in[17];

    float* x = (float*)d_out;   // running residual, f32

    char* ws = (char*)d_ws;
    const size_t szMD  = (size_t)MTOT * DMODEL * sizeof(__bf16);
    const size_t szMF  = (size_t)MTOT * DFFN  * sizeof(__bf16);
    size_t off = 0;
    __bf16* hB   = (__bf16*)(ws + off); off += szMD;
    __bf16* qB   = (__bf16*)(ws + off); off += szMD;
    __bf16* kB   = (__bf16*)(ws + off); off += szMD;
    __bf16* vB   = (__bf16*)(ws + off); off += szMD;
    __bf16* ctxB = (__bf16*)(ws + off); off += szMD;
    __bf16* gB   = (__bf16*)(ws + off); off += szMF;
    __bf16* WqT  = (__bf16*)(ws + off); off += (size_t)DMODEL * DMODEL * 2;
    __bf16* WkT  = (__bf16*)(ws + off); off += (size_t)DMODEL * DMODEL * 2;
    __bf16* WvT  = (__bf16*)(ws + off); off += (size_t)DMODEL * DMODEL * 2;
    __bf16* WoT  = (__bf16*)(ws + off); off += (size_t)DMODEL * DMODEL * 2;
    __bf16* WpT  = (__bf16*)(ws + off); off += (size_t)(2 * DFFN) * DMODEL * 2;
    __bf16* WfT  = (__bf16*)(ws + off); off += (size_t)DMODEL * DFFN * 2;

    {
        size_t n = (size_t)MTOT * DMODEL;
        copy_f32_kernel<<<(unsigned)((n + 255) / 256), 256, 0, stream>>>(x_in, x, n);
    }

    const dim3 gemmBlk(256);
    const dim3 gridDD(DMODEL / 64, MTOT / 128);   // 8 x 256
    const dim3 gridFF(DFFN / 64, MTOT / 128);     // 30 x 256

    for (int l = 0; l < LAYERS; ++l) {
        const size_t oDD = (size_t)l * DMODEL * DMODEL;
        const size_t oD  = (size_t)l * DMODEL;
        const size_t oP  = (size_t)l * DMODEL * 2 * DFFN;
        const size_t oF  = (size_t)l * DFFN * DMODEL;
        const size_t oT  = (size_t)l * NREL * NREL * NHEAD;

        {
            unsigned nDD = (DMODEL * DMODEL + 255) / 256;
            transpose_bf16_kernel<<<nDD, 256, 0, stream>>>(Wq + oDD, WqT, DMODEL, DMODEL);
            transpose_bf16_kernel<<<nDD, 256, 0, stream>>>(Wk + oDD, WkT, DMODEL, DMODEL);
            transpose_bf16_kernel<<<nDD, 256, 0, stream>>>(Wv + oDD, WvT, DMODEL, DMODEL);
            transpose_bf16_kernel<<<nDD, 256, 0, stream>>>(Wo + oDD, WoT, DMODEL, DMODEL);
            unsigned nP = ((unsigned)(DMODEL * 2 * DFFN) + 255) / 256;
            transpose_bf16_kernel<<<nP, 256, 0, stream>>>(Wp + oP, WpT, DMODEL, 2 * DFFN);
            unsigned nF = ((unsigned)(DFFN * DMODEL) + 255) / 256;
            transpose_bf16_kernel<<<nF, 256, 0, stream>>>(Wf + oF, WfT, DFFN, DMODEL);
        }

        layernorm_kernel<<<MTOT, 128, 0, stream>>>(x, ln1w + oD, ln1b + oD, hB);

        gemm_wmma_kernel<0><<<gridDD, gemmBlk, 0, stream>>>(hB, DMODEL, WqT, DMODEL,
                                                            bq + oD, qB, nullptr, DMODEL, DMODEL);
        gemm_wmma_kernel<0><<<gridDD, gemmBlk, 0, stream>>>(hB, DMODEL, WkT, DMODEL,
                                                            bk + oD, kB, nullptr, DMODEL, DMODEL);
        gemm_wmma_kernel<0><<<gridDD, gemmBlk, 0, stream>>>(hB, DMODEL, WvT, DMODEL,
                                                            bv + oD, vB, nullptr, DMODEL, DMODEL);

        attention_kernel<<<BATCH * NHEAD, 128, 0, stream>>>(qB, kB, vB, tab + oT, ctxB);

        gemm_wmma_kernel<1><<<gridDD, gemmBlk, 0, stream>>>(ctxB, DMODEL, WoT, DMODEL,
                                                            bo + oD, nullptr, x, DMODEL, DMODEL);

        layernorm_kernel<<<MTOT, 128, 0, stream>>>(x, ln2w + oD, ln2b + oD, hB);

        ffn_gate_kernel<<<gridFF, gemmBlk, 0, stream>>>(hB, WpT, bp + (size_t)l * 2 * DFFN, gB);

        gemm_wmma_kernel<1><<<gridDD, gemmBlk, 0, stream>>>(gB, DFFN, WfT, DFFN,
                                                            bfb + oD, nullptr, x, DMODEL, DFFN);
    }
}
